// MultiHeadAttention_82849919140503
// MI455X (gfx1250) — matrix operand order repaired and checked by reading, not compiled, not run on hardware
//
#include <hip/hip_runtime.h>
#include <math.h>

// Sizes fixed by the reference
#define BB 16
#define LL 4096
#define DD 1024
#define HH 16
#define DHH 64
#define LNEPS 1e-5f

typedef __attribute__((ext_vector_type(16))) __bf16    v16bf;
typedef __attribute__((ext_vector_type(8)))  float     v8f;
typedef __attribute__((ext_vector_type(8)))  unsigned  v8u;

// Pack two f32 into one dword of 2x bf16 (truncation). Lowers to v_perm_b32.
static __device__ __forceinline__ unsigned pack2bf(float lo, float hi) {
  unsigned a = __builtin_bit_cast(unsigned, lo);
  unsigned b = __builtin_bit_cast(unsigned, hi);
  return (a >> 16) | (b & 0xffff0000u);
}

// ---------------- small precompute kernels ----------------

// qf[b,d] = query[b,:] @ Wq[:,d] + bq[d]
__global__ void qproj_kernel(const float* __restrict__ query, const float* __restrict__ Wq,
                             const float* __restrict__ bq, float* __restrict__ qf) {
  int idx = blockIdx.x * blockDim.x + threadIdx.x;   // B*D
  int b = idx / DD, d = idx % DD;
  const float* qrow = query + b * DD;
  float acc = bq[d];
  #pragma unroll 4
  for (int e = 0; e < DD; ++e) acc += qrow[e] * Wq[(size_t)e * DD + d];
  qf[idx] = acc;
}

// qk[b,h,e] = sum_c qf[b, h*64+c] * Wk[e, h*64+c]
__global__ void qkvec_kernel(const float* __restrict__ qf, const float* __restrict__ Wk,
                             float* __restrict__ qk) {
  int idx = blockIdx.x * blockDim.x + threadIdx.x;   // B*H*D
  int e = idx % DD;
  int bh = idx / DD;
  int h = bh % HH, b = bh / HH;
  const float* q = qf + b * DD + h * DHH;
  const float* w = Wk + (size_t)e * DD + h * DHH;
  float acc = 0.f;
  #pragma unroll
  for (int c = 0; c < DHH; ++c) acc += q[c] * w[c];
  qk[idx] = acc;
}

// qbk[b,h] = q_h . bk_h  (bias term of the scores)
__global__ void qbk_kernel(const float* __restrict__ qf, const float* __restrict__ bk,
                           float* __restrict__ qbk) {
  int t = threadIdx.x;                                // 256 == B*H
  int h = t % HH, b = t / HH;
  const float* q = qf + b * DD + h * DHH;
  const float* bb = bk + h * DHH;
  float acc = 0.f;
  #pragma unroll
  for (int c = 0; c < DHH; ++c) acc += q[c] * bb[c];
  qbk[t] = acc;
}

// ---------------- scores: 16(heads) x 16(l-tile) WMMA over K=D ----------------
// scores_b = qk_b (16xD) @ key_b^T, streamed once over key (256 MB total)
__global__ void scores_kernel(const float* __restrict__ qk, const float* __restrict__ qbk,
                              const float* __restrict__ key, float* __restrict__ outp) {
  int b    = blockIdx.x;
  int wave = threadIdx.x >> 5;
  int lane = threadIdx.x & 31;
  int l0   = blockIdx.y * 64 + wave * 16;
  int mlo  = lane & 15;
  int khalf = lane >> 4;

  const float* arow = qk + (size_t)(b * HH + mlo) * DD;          // A row = head
  const float* krow = key + ((size_t)b * LL + l0 + mlo) * DD;    // B col  = l
  v8f c = {};
  for (int kk = 0; kk < DD; kk += 32) {
    __builtin_prefetch(krow + kk + 256, 0, 0);                   // stream-ahead 1KB
    v8u au, bu;
    #pragma unroll
    for (int i = 0; i < 8; ++i) {
      int ka = kk + ((i >> 2) << 4) + (khalf << 3) + ((i & 3) << 1);   // A 16-bit layout
      float2 pa = *(const float2*)(arow + ka);
      au[i] = pack2bf(pa.x, pa.y);
      int kb = kk + ((i >> 2) << 4) + (khalf << 3) + ((i & 3) << 1);   // B 16-bit layout
      float2 pb = *(const float2*)(krow + kb);
      bu[i] = pack2bf(pb.x, pb.y);
    }
    v16bf a  = __builtin_bit_cast(v16bf, au);
    v16bf bm = __builtin_bit_cast(v16bf, bu);
    c = __builtin_amdgcn_wmma_f32_16x16x32_bf16(false, a, false, bm, (short)0, c, false, false);
  }
  float* attn = outp + BB * DD;                                   // raw scores -> attn region
  #pragma unroll
  for (int i = 0; i < 8; ++i) {
    int m = i + (khalf << 3);
    attn[(size_t)(b * HH + m) * LL + l0 + mlo] = (c[i] + qbk[b * HH + m]) * 0.125f;
  }
}

// ---------------- in-place softmax over L for each (b,h) row ----------------
__global__ void softmax_kernel(float* __restrict__ outp) {
  int r = blockIdx.x;                        // B*H rows
  float* row = outp + BB * DD + (size_t)r * LL;
  int tid = threadIdx.x;
  __shared__ float red[256];
  float v[LL / 256];
  float mx = -1e30f;
  #pragma unroll
  for (int i = 0; i < LL / 256; ++i) { v[i] = row[tid + i * 256]; mx = fmaxf(mx, v[i]); }
  red[tid] = mx; __syncthreads();
  for (int s = 128; s > 0; s >>= 1) {
    if (tid < s) red[tid] = fmaxf(red[tid], red[tid + s]);
    __syncthreads();
  }
  mx = red[0]; __syncthreads();
  float sum = 0.f;
  #pragma unroll
  for (int i = 0; i < LL / 256; ++i) { v[i] = __expf(v[i] - mx); sum += v[i]; }
  red[tid] = sum; __syncthreads();
  for (int s = 128; s > 0; s >>= 1) {
    if (tid < s) red[tid] += red[tid + s];
    __syncthreads();
  }
  float inv = 1.0f / red[0];
  #pragma unroll
  for (int i = 0; i < LL / 256; ++i) row[tid + i * 256] = v[i] * inv;
}

// ---------------- t_b = attn_b (16xL) @ value_b (LxD), WMMA, 4 waves split K ----------------
__global__ void tred_kernel(const float* __restrict__ outp, const float* __restrict__ value,
                            float* __restrict__ t) {
  int b    = blockIdx.x;
  int n0   = blockIdx.y * 16;
  int wave = threadIdx.x >> 5;
  int lane = threadIdx.x & 31;
  int mlo  = lane & 15;
  int khalf = lane >> 4;

  const float* arow = outp + BB * DD + (size_t)(b * HH + mlo) * LL;  // attn row, K = l
  const float* vbase = value + (size_t)b * LL * DD + n0 + mlo;       // column n = mlo
  v8f c = {};
  int kbase = wave * (LL / 4);
  for (int kk = kbase; kk < kbase + LL / 4; kk += 32) {
    v8u au, bu;
    #pragma unroll
    for (int i = 0; i < 8; ++i) {
      int ka = kk + ((i >> 2) << 4) + (khalf << 3) + ((i & 3) << 1);
      float2 pa = *(const float2*)(arow + ka);
      au[i] = pack2bf(pa.x, pa.y);
      int kb = kk + ((i >> 2) << 4) + (khalf << 3) + ((i & 3) << 1); // rows kb, kb+1 of value
      const float* vp = vbase + (size_t)kb * DD;
      __builtin_prefetch(vp + 32 * DD, 0, 0);                        // stream-ahead 32 rows
      bu[i] = pack2bf(vp[0], vp[DD]);
    }
    v16bf a  = __builtin_bit_cast(v16bf, au);
    v16bf bm = __builtin_bit_cast(v16bf, bu);
    c = __builtin_amdgcn_wmma_f32_16x16x32_bf16(false, a, false, bm, (short)0, c, false, false);
  }
  __shared__ float red[4][32][8];
  #pragma unroll
  for (int i = 0; i < 8; ++i) red[wave][lane][i] = c[i];
  __syncthreads();
  if (threadIdx.x < 32) {
    #pragma unroll
    for (int i = 0; i < 8; ++i) {
      float s = red[0][lane][i] + red[1][lane][i] + red[2][lane][i] + red[3][lane][i];
      int m = i + (khalf << 3);
      t[(size_t)(b * HH + m) * DD + n0 + mlo] = s;
    }
  }
}

// ---------------- ctx[b,c] = t[b, h(c), :] @ Wv[:, c] + bv[c] ----------------
__global__ void ctx_kernel(const float* __restrict__ t, const float* __restrict__ Wv,
                           const float* __restrict__ bv, float* __restrict__ ctx) {
  int idx = blockIdx.x * blockDim.x + threadIdx.x;   // B*D
  int b = idx / DD, cd = idx % DD;
  int h = cd >> 6;
  const float* trow = t + (size_t)(b * HH + h) * DD;
  float acc = bv[cd];
  #pragma unroll 4
  for (int e = 0; e < DD; ++e) acc += trow[e] * Wv[(size_t)e * DD + cd];
  ctx[idx] = acc;
}

// ---------------- out = LN(ctx @ Wo + bo) * gamma + beta ----------------
__global__ void outln_kernel(const float* __restrict__ ctx, const float* __restrict__ Wo,
                             const float* __restrict__ bo, const float* __restrict__ gamma,
                             const float* __restrict__ beta, float* __restrict__ outp) {
  int b = blockIdx.x;
  int tid = threadIdx.x;   // 256 threads; each computes 4 outputs
  __shared__ float sctx[DD];
  __shared__ float rsum[256], rsq[256];
  for (int cd = tid; cd < DD; cd += 256) sctx[cd] = ctx[b * DD + cd];
  __syncthreads();
  float o[4];
  #pragma unroll
  for (int v = 0; v < 4; ++v) {
    int j = tid + v * 256;
    float acc = bo[j];
    for (int cd = 0; cd < DD; ++cd) acc += sctx[cd] * Wo[(size_t)cd * DD + j];
    o[v] = acc;
  }
  float s = 0.f, sq = 0.f;
  #pragma unroll
  for (int v = 0; v < 4; ++v) { s += o[v]; sq += o[v] * o[v]; }
  rsum[tid] = s; rsq[tid] = sq; __syncthreads();
  for (int st = 128; st > 0; st >>= 1) {
    if (tid < st) { rsum[tid] += rsum[tid + st]; rsq[tid] += rsq[tid + st]; }
    __syncthreads();
  }
  float mean = rsum[0] * (1.0f / DD);
  float var  = rsq[0] * (1.0f / DD) - mean * mean;
  float inv  = rsqrtf(var + LNEPS);
  #pragma unroll
  for (int v = 0; v < 4; ++v) {
    int j = tid + v * 256;
    outp[b * DD + j] = (o[v] - mean) * inv * gamma[j] + beta[j];
  }
}

extern "C" void kernel_launch(void* const* d_in, const int* in_sizes, int n_in,
                              void* d_out, int out_size, void* d_ws, size_t ws_size,
                              hipStream_t stream) {
  const float* query = (const float*)d_in[0];
  const float* key   = (const float*)d_in[1];
  const float* value = (const float*)d_in[2];
  const float* Wq    = (const float*)d_in[3];
  const float* bq    = (const float*)d_in[4];
  const float* Wk    = (const float*)d_in[5];
  const float* bk    = (const float*)d_in[6];
  const float* Wv    = (const float*)d_in[7];
  const float* bv    = (const float*)d_in[8];
  const float* Wo    = (const float*)d_in[9];
  const float* bo    = (const float*)d_in[10];
  const float* gamma = (const float*)d_in[11];
  const float* beta  = (const float*)d_in[12];

  float* outp = (float*)d_out;                 // [0,B*D) = out, [B*D, B*D+B*H*L) = attn
  float* ws = (float*)d_ws;
  float* qf   = ws;                            // B*D
  float* qk   = qf + BB * DD;                  // B*H*D
  float* qbkp = qk + BB * HH * DD;             // B*H
  float* t    = qbkp + BB * HH;                // B*H*D
  float* ctx  = t + BB * HH * DD;              // B*D

  qproj_kernel  <<<BB * DD / 256, 256, 0, stream>>>(query, Wq, bq, qf);
  qkvec_kernel  <<<BB * HH * DD / 256, 256, 0, stream>>>(qf, Wk, qk);
  qbk_kernel    <<<1, 256, 0, stream>>>(qf, bk, qbkp);
  scores_kernel <<<dim3(BB, LL / 64), 128, 0, stream>>>(qk, qbkp, key, outp);
  softmax_kernel<<<BB * HH, 256, 0, stream>>>(outp);
  tred_kernel   <<<dim3(BB, DD / 16), 128, 0, stream>>>(outp, value, t);
  ctx_kernel    <<<BB * DD / 256, 256, 0, stream>>>(t, Wv, bv, ctx);
  outln_kernel  <<<BB, 256, 0, stream>>>(ctx, Wo, bo, gamma, beta, outp);
}
